// KPConv_49555332661489
// MI455X (gfx1250) — matrix operand order repaired and checked by reading, not compiled, not run on hardware
//
#include <hip/hip_runtime.h>

// ---------------------------------------------------------------------------
// KPConv fused for gfx1250 (MI455X, wave32, WMMA).
//   stage 1: per-query [16x32]x[32x64] influence matmul  -> v_wmma_f32_16x16x32_f16
//   stage 2: [N x 960] x [960 x 128] GEMM, 32-query M-tile per block,
//            2 accumulators/wave so each B fragment feeds 2 WMMAs.
// f16 inputs / f32 accumulation; gathers and Wt are L2-resident.
// ---------------------------------------------------------------------------

typedef _Float16 v16h __attribute__((ext_vector_type(16)));
typedef float    v8f  __attribute__((ext_vector_type(8)));

union F16Frag { v16h v; uint4 q[2]; };

#define HNB   32          // neighbors per query
#define KP    15          // kernel points (padded to 16 in stage-1 A)
#define CIN   64
#define COUT  128
#define KDIM  960         // 15*64 contraction dim of stage 2
#define QTILE 32          // queries per block (stage-2 M tile)
#define A2S   968         // padded LDS row stride (halfs), multiple of 8 -> 16B-aligned b128
#define XTS   40          // xT row stride (halfs), multiple of 8
#define A1S   40          // A1 row stride (halfs)

#define WF_HALFS   (QTILE * A2S)         // 30976 halfs (61952 B) shared wf panel
#define A1_HALFS   (16 * A1S)            // 640
#define XT_HALFS   (64 * XTS)            // 2560
#define WAVE_HALFS (A1_HALFS + XT_HALFS) // 3200 per-wave stage-1 scratch
#define LDS_BYTES  ((WF_HALFS + 8 * WAVE_HALFS) * 2)   // 113152 B

__device__ __forceinline__ unsigned int pack2h(float a, float b) {
    union { _Float16 h[2]; unsigned int u; } p;
    p.h[0] = (_Float16)a; p.h[1] = (_Float16)b;
    return p.u;
}

// Repack weights [15][64][128] f32  ->  Wt[o][k*64+c] f16 (B-transposed for WMMA)
__global__ void kpconv_prep_w(const float* __restrict__ w, _Float16* __restrict__ wt) {
    int t = blockIdx.x * blockDim.x + threadIdx.x;
    if (t < KDIM * COUT) {
        int o  = t & (COUT - 1);
        int kc = t >> 7;
        wt[o * KDIM + kc] = (_Float16)w[kc * COUT + o];
    }
}

__global__ void __launch_bounds__(256)
kpconv_fused(const float* __restrict__ q_pts, const float* __restrict__ s_pts,
             const long long* __restrict__ inds, const float* __restrict__ x,
             const float* __restrict__ kpts, const _Float16* __restrict__ wt,
             const float* __restrict__ bias, float* __restrict__ out, int N)
{
    extern __shared__ _Float16 smem[];
    _Float16* wfL = smem;                                   // [QTILE][A2S] stage-2 A panel

    const int lane = threadIdx.x & 31;
    const int wave = threadIdx.x >> 5;
    const int lo   = lane & 15;
    const bool hi  = lane >= 16;

    _Float16* A1 = smem + WF_HALFS + wave * WAVE_HALFS;     // [16kp][A1S]
    _Float16* xT = A1 + A1_HALFS;                           // [64c][XTS]

    const int   qbase   = blockIdx.x * QTILE;
    const float inv_ext = 1.0f / 0.072f;

    // ---------------- stage 1: four queries per wave ----------------
    for (int qq = 0; qq < 4; ++qq) {
        const int q  = wave * 4 + qq;                        // block-local row 0..31
        int qg = qbase + q; if (qg >= N) qg = N - 1;
        const long long* nrow = inds + (long long)qg * HNB;

        // influence weights: lane == neighbor h
        {
            long long idx = nrow[lane];
            float qx = q_pts[qg * 3 + 0], qy = q_pts[qg * 3 + 1], qz = q_pts[qg * 3 + 2];
            const float* sp = s_pts + idx * 3;
            float rx = sp[0] - qx, ry = sp[1] - qy, rz = sp[2] - qz;
            #pragma unroll
            for (int k = 0; k < KP; ++k) {
                float dx = rx - kpts[k * 3 + 0];
                float dy = ry - kpts[k * 3 + 1];
                float dz = rz - kpts[k * 3 + 2];
                float d  = __builtin_sqrtf(dx * dx + dy * dy + dz * dz);
                float w  = 1.0f - d * inv_ext;
                w = w > 0.0f ? w : 0.0f;
                A1[k * A1S + lane] = (_Float16)w;
            }
            A1[15 * A1S + lane] = (_Float16)0.0f;            // pad row
        }

        // gather features, transposed + f16-packed: lane -> (neighbor pair p, c-half b)
        {
            const int p = lane & 15, b = lane >> 4;
            long long i0 = nrow[2 * p], i1 = nrow[2 * p + 1];
            const float4* x0 = (const float4*)(x + i0 * CIN + 32 * b);
            const float4* x1 = (const float4*)(x + i1 * CIN + 32 * b);
            #pragma unroll
            for (int c4 = 0; c4 < 8; ++c4) {
                float4 a  = x0[c4];
                float4 bb = x1[c4];
                int cb = 32 * b + c4 * 4;
                *(unsigned int*)&xT[(cb + 0) * XTS + 2 * p] = pack2h(a.x, bb.x);
                *(unsigned int*)&xT[(cb + 1) * XTS + 2 * p] = pack2h(a.y, bb.y);
                *(unsigned int*)&xT[(cb + 2) * XTS + 2 * p] = pack2h(a.z, bb.z);
                *(unsigned int*)&xT[(cb + 3) * XTS + 2 * p] = pack2h(a.w, bb.w);
            }
        }
        __syncthreads();

        // wf[kp][c] = A1(16x32) * X(32x64) : 4 WMMAs per query
        F16Frag af;
        {
            // A 16-bit 16x32 layout: lanes<16 hold K{0..7,16..23}, lanes>=16 K{8..15,24..31}
            const _Float16* pa = A1 + lo * A1S + (hi ? 8 : 0);
            af.q[0] = *(const uint4*)pa;
            af.q[1] = *(const uint4*)(pa + 16);
        }
        const int kb = hi ? 8 : 0;
        #pragma unroll
        for (int ct = 0; ct < 4; ++ct) {
            F16Frag bf;
            // B layout: lane = column c, lanes<16 hold K{0..7,16..23}, lanes>=16 K{8..15,24..31}
            const _Float16* pb = xT + (ct * 16 + lo) * XTS + (hi ? 8 : 0);
            bf.q[0] = *(const uint4*)pb;
            bf.q[1] = *(const uint4*)(pb + 16);
            v8f d = {};
            d = __builtin_amdgcn_wmma_f32_16x16x32_f16(false, af.v, false, bf.v,
                                                       (short)0, d, false, false);
            #pragma unroll
            for (int v = 0; v < 8; ++v) {
                int kp = kb + v;                             // D: VGPR v -> row kb+v
                if (kp < KP)
                    wfL[q * A2S + kp * CIN + ct * 16 + lo] = (_Float16)d[v];
            }
        }
        __syncthreads();
    }

    // ------- stage 2: two [16q x 16o] tiles per wave, shared B fragment, K = 960 -------
    const int otile = wave * 16;
    const _Float16* wbase = wt + (long long)(otile + lo) * KDIM;

    v8f acc0 = {};                                           // rows 0..15
    v8f acc1 = {};                                           // rows 16..31
    #pragma unroll 2
    for (int kk = 0; kk < 30; ++kk) {
        const int k0 = kk * 32;
        __builtin_prefetch(wbase + k0 + 256, 0, 1);          // global_prefetch_b8

        F16Frag b2;                                          // from L2-resident Wt
        const _Float16* pb = wbase + k0 + (hi ? 8 : 0);
        b2.q[0] = *(const uint4*)pb;
        b2.q[1] = *(const uint4*)(pb + 16);

        F16Frag a0, a1;                                      // from LDS wf panel
        const _Float16* pa0 = wfL + lo * A2S + k0 + (hi ? 8 : 0);
        a0.q[0] = *(const uint4*)pa0;
        a0.q[1] = *(const uint4*)(pa0 + 16);
        const _Float16* pa1 = pa0 + 16 * A2S;
        a1.q[0] = *(const uint4*)pa1;
        a1.q[1] = *(const uint4*)(pa1 + 16);

        acc0 = __builtin_amdgcn_wmma_f32_16x16x32_f16(false, a0.v, false, b2.v,
                                                      (short)0, acc0, false, false);
        acc1 = __builtin_amdgcn_wmma_f32_16x16x32_f16(false, a1.v, false, b2.v,
                                                      (short)0, acc1, false, false);
    }

    float bv = bias[otile + lo];
    #pragma unroll
    for (int v = 0; v < 8; ++v) {
        int qrow = (hi ? 8 : 0) + v;                         // D: VGPR v -> row
        int qg0  = qbase + qrow;
        int qg1  = qbase + 16 + qrow;
        if (qg0 < N) out[(long long)qg0 * COUT + otile + lo] = acc0[v] + bv;
        if (qg1 < N) out[(long long)qg1 * COUT + otile + lo] = acc1[v] + bv;
    }
}

extern "C" void kernel_launch(void* const* d_in, const int* in_sizes, int n_in,
                              void* d_out, int out_size, void* d_ws, size_t ws_size,
                              hipStream_t stream) {
    const float*     q_pts = (const float*)d_in[0];
    const float*     s_pts = (const float*)d_in[1];
    const long long* inds  = (const long long*)d_in[2];
    const float*     x     = (const float*)d_in[3];
    const float*     kpts  = (const float*)d_in[4];
    const float*     w     = (const float*)d_in[5];
    const float*     bias  = (const float*)d_in[6];
    float*           out   = (float*)d_out;
    _Float16*        wt    = (_Float16*)d_ws;

    if (ws_size < (size_t)KDIM * COUT * sizeof(_Float16)) return;

    int N = in_sizes[0] / 3;

    kpconv_prep_w<<<(KDIM * COUT + 255) / 256, 256, 0, stream>>>(w, wt);

    int nblocks = (N + QTILE - 1) / QTILE;
    kpconv_fused<<<nblocks, 256, LDS_BYTES, stream>>>(q_pts, s_pts, inds, x, kpts,
                                                      wt, bias, out, N);
}
